// GCN_5385888989806
// MI455X (gfx1250) — compile-verified, matrix operand order checked by reading, not run on hardware
//
#include <hip/hip_runtime.h>
#include <hip/hip_bf16.h>

typedef __attribute__((ext_vector_type(16))) _Float16 v16h;
typedef __attribute__((ext_vector_type(8)))  float    v8f;

#define T_HIST 3
#define F_IN   5
#define HDIM   128
#define CDIM   5

// ---------------------------------------------------------------- utilities
__global__ void fill_zero_kernel(float* __restrict__ p, int n) {
    int i = blockIdx.x * blockDim.x + threadIdx.x;
    if (i < n) p[i] = 0.0f;
}

__global__ void deg_kernel(const int* __restrict__ dst, float* __restrict__ deg, int e) {
    int i = blockIdx.x * blockDim.x + threadIdx.x;
    if (i < e) atomicAdd(&deg[dst[i]], 1.0f);
}

__global__ void dinv_kernel(float* __restrict__ deg, int n) {
    int i = blockIdx.x * blockDim.x + threadIdx.x;
    if (i < n) deg[i] = rsqrtf(deg[i] + 1.0f);
}

// ---------------------------------------------------------------- layer-1 GEMM (K=5) fused with agg seed
// xw1 = x_last @ W1 ; agg_seed = dinv^2 * xw1 + b1  (self-loop + bias pre-seeded)
__global__ void xw1_seed_kernel(const float* __restrict__ x, const float* __restrict__ W1,
                                const float* __restrict__ b1, const float* __restrict__ dinv,
                                float* __restrict__ xw1, float* __restrict__ agg_seed, int n) {
    int idx = blockIdx.x * blockDim.x + threadIdx.x;        // over N*H
    if (idx >= n * HDIM) return;
    int row = idx >> 7;
    int h   = idx & (HDIM - 1);
    const float* xr = x + (size_t)row * (T_HIST * F_IN) + (T_HIST - 1) * F_IN;
    float s = 0.0f;
#pragma unroll
    for (int f = 0; f < F_IN; ++f) s = fmaf(xr[f], W1[f * HDIM + h], s);
    xw1[idx] = s;
    float di = dinv[row];
    agg_seed[idx] = di * di * s + b1[h];
}

// ---------------------------------------------------------------- edge scatter: one wave per edge, float4 lanes
__global__ __launch_bounds__(256) void scatter_kernel(const int* __restrict__ src, const int* __restrict__ dst,
                                                      const float* __restrict__ dinv, const float* __restrict__ xw,
                                                      float* __restrict__ agg, int e) {
    int edge = (blockIdx.x * blockDim.x + threadIdx.x) >> 5;
    int lane = threadIdx.x & 31;
    if (edge >= e) return;
    int s = src[edge], d = dst[edge];
    float norm = dinv[s] * dinv[d];
    const float4* xr = (const float4*)(xw + (size_t)s * HDIM);
    float4 v = xr[lane];                                     // 512B coalesced gather per wave
    float* ag = agg + (size_t)d * HDIM + lane * 4;
    atomicAdd(ag + 0, norm * v.x);
    atomicAdd(ag + 1, norm * v.y);
    atomicAdd(ag + 2, norm * v.z);
    atomicAdd(ag + 3, norm * v.w);
}

// ---------------------------------------------------------------- layer-2 GEMM: xw2 = relu(pre1) @ W2 (WMMA f16->f32)
// block = 8 waves, 16 rows x 128 cols; ALSO seeds agg2 = dinv^2*xw2 + b2 in place of pre1 rows
// (hazard-free: each block reads/writes only its own 16 rows, reads complete at staging).
__global__ __launch_bounds__(256) void gemm2_wmma_kernel(const float* __restrict__ pre1,   // in: layer-1 pre-act (bufB)
                                                         const float* __restrict__ W2,
                                                         const float* __restrict__ b2,
                                                         const float* __restrict__ dinv,
                                                         float* __restrict__ xw2,          // out: bufA
                                                         float* __restrict__ agg_seed,     // out: bufB (in place)
                                                         int nrows) {
    __shared__ _Float16 aS[16][HDIM];        // relu(A) tile [m][k]       4 KB
    __shared__ _Float16 bS[HDIM][HDIM];      // W2 transposed [n][k]     32 KB
    __shared__ float    dS[16];              // dinv for this row block

    const int tid       = threadIdx.x;
    const int block_row = blockIdx.x * 16;

    for (int i = tid; i < HDIM * HDIM; i += 256) {
        int k = i >> 7;
        int n = i & (HDIM - 1);
        bS[n][k] = (_Float16)W2[i];
    }
    for (int i = tid; i < 16 * HDIM; i += 256) {
        int m = i >> 7;
        int k = i & (HDIM - 1);
        int row = block_row + m;
        float v = (row < nrows) ? fmaxf(pre1[(size_t)row * HDIM + k], 0.0f) : 0.0f;  // fused ReLU
        aS[m][k] = (_Float16)v;
    }
    if (tid < 16) {
        int row = block_row + tid;
        dS[tid] = (row < nrows) ? dinv[row] : 0.0f;
    }
    __syncthreads();

    const int wave  = tid >> 5;
    const int lane  = tid & 31;
    const int ncol  = wave * 16;
    const int mrow  = lane & 15;
    const int khalf = (lane >> 4) * 8;

    v8f acc = {};
    for (int kk = 0; kk < HDIM; kk += 32) {
        v16h a, b;
#pragma unroll
        for (int j = 0; j < 16; j += 2) {
            int K = (j & 7) + ((j >> 3) << 4) + khalf + kk;   // ISA 16-bit A/B fragment K map
            a[j]     = aS[mrow][K];
            a[j + 1] = aS[mrow][K + 1];
            b[j]     = bS[ncol + mrow][K];
            b[j + 1] = bS[ncol + mrow][K + 1];
        }
        acc = __builtin_amdgcn_wmma_f32_16x16x32_f16(false, a, false, b,
                                                     (short)0, acc, false, false);
    }

    const int col  = ncol + (lane & 15);
    const float bc = b2[col];
    if (block_row + 16 <= nrows) {           // fast path: no tail (N % 16 == 0 in practice)
#pragma unroll
        for (int v = 0; v < 8; ++v) {
            int r  = v + ((lane >> 4) << 3);
            int row = block_row + r;
            float val = acc[v];
            float di  = dS[r];
            xw2[(size_t)row * HDIM + col]      = val;
            agg_seed[(size_t)row * HDIM + col] = di * di * val + bc;
        }
    } else {
#pragma unroll
        for (int v = 0; v < 8; ++v) {
            int r  = v + ((lane >> 4) << 3);
            int row = block_row + r;
            if (row < nrows) {
                float val = acc[v];
                float di  = dS[r];
                xw2[(size_t)row * HDIM + col]      = val;
                agg_seed[(size_t)row * HDIM + col] = di * di * val + bc;
            }
        }
    }
}

// ---------------------------------------------------------------- output head: wave per node, ReLU fused, softmax
__global__ __launch_bounds__(256) void out_kernel(const float* __restrict__ pre2,
                                                  const float* __restrict__ Wout,
                                                  const float* __restrict__ bout,
                                                  float* __restrict__ out, int n) {
    int node = (blockIdx.x * blockDim.x + threadIdx.x) >> 5;
    int lane = threadIdx.x & 31;
    if (node >= n) return;

    const float4* hr = (const float4*)(pre2 + (size_t)node * HDIM);
    float4 hv = hr[lane];                                    // 512B coalesced per wave
    hv.x = fmaxf(hv.x, 0.0f);
    hv.y = fmaxf(hv.y, 0.0f);
    hv.z = fmaxf(hv.z, 0.0f);
    hv.w = fmaxf(hv.w, 0.0f);

    int h0 = lane * 4;
    float acc[CDIM];
#pragma unroll
    for (int c = 0; c < CDIM; ++c) {
        acc[c] = hv.x * Wout[(h0 + 0) * CDIM + c]
               + hv.y * Wout[(h0 + 1) * CDIM + c]
               + hv.z * Wout[(h0 + 2) * CDIM + c]
               + hv.w * Wout[(h0 + 3) * CDIM + c];
    }
#pragma unroll
    for (int off = 16; off > 0; off >>= 1) {
#pragma unroll
        for (int c = 0; c < CDIM; ++c) acc[c] += __shfl_xor(acc[c], off, 32);
    }
    if (lane == 0) {
#pragma unroll
        for (int c = 0; c < CDIM; ++c) acc[c] += bout[c];
        float m = acc[0];
#pragma unroll
        for (int c = 1; c < CDIM; ++c) m = fmaxf(m, acc[c]);
        float sum = 0.0f;
#pragma unroll
        for (int c = 0; c < CDIM; ++c) { acc[c] = __expf(acc[c] - m); sum += acc[c]; }
        float inv = 1.0f / sum;
#pragma unroll
        for (int c = 0; c < CDIM; ++c) out[(size_t)node * CDIM + c] = acc[c] * inv;
    }
}

// ---------------------------------------------------------------- driver
extern "C" void kernel_launch(void* const* d_in, const int* in_sizes, int n_in,
                              void* d_out, int out_size, void* d_ws, size_t ws_size,
                              hipStream_t stream) {
    const float* x    = (const float*)d_in[0];
    const int*   eidx = (const int*)  d_in[1];
    const float* W1   = (const float*)d_in[2];
    const float* b1   = (const float*)d_in[3];
    const float* W2   = (const float*)d_in[4];
    const float* b2   = (const float*)d_in[5];
    const float* Wout = (const float*)d_in[6];
    const float* bout = (const float*)d_in[7];
    float*       out  = (float*)d_out;

    const int N = in_sizes[0] / (T_HIST * F_IN);
    const int E = in_sizes[1] / 2;
    const int* src = eidx;
    const int* dst = eidx + E;

    // workspace layout: dinv | bufA (N*H) | bufB (N*H)
    float* dinv = (float*)d_ws;
    float* bufA = dinv + ((N + 63) & ~63);
    float* bufB = bufA + (size_t)N * HDIM;

    const int TB = 256;
    const int NH = N * HDIM;
    const int NH_blocks   = (NH + TB - 1) / TB;
    const int edge_blocks = ((E * 32) + TB - 1) / TB;        // one wave per edge
    const int node_blocks = ((N * 32) + TB - 1) / TB;        // one wave per node

    // degree normalization: dinv = rsqrt(deg + 1)
    fill_zero_kernel<<<(N + TB - 1) / TB, TB, 0, stream>>>(dinv, N);
    deg_kernel<<<(E + TB - 1) / TB, TB, 0, stream>>>(dst, dinv, E);
    dinv_kernel<<<(N + TB - 1) / TB, TB, 0, stream>>>(dinv, N);

    // ---- layer 1: xw1 -> bufA, agg seeded with self-loop+bias -> bufB
    xw1_seed_kernel<<<NH_blocks, TB, 0, stream>>>(x, W1, b1, dinv, bufA, bufB, N);
    scatter_kernel<<<edge_blocks, TB, 0, stream>>>(src, dst, dinv, bufA, bufB, E);
    // bufB now = pre-activation of layer 1 (ReLU deferred into gemm2 staging)

    // ---- layer 2: WMMA GEMM with fused ReLU(A) and fused agg2 seed (bufB in place)
    gemm2_wmma_kernel<<<(N + 15) / 16, 256, 0, stream>>>(bufB, W2, b2, dinv, bufA, bufB, N);
    scatter_kernel<<<edge_blocks, TB, 0, stream>>>(src, dst, dinv, bufA, bufB, E);
    // bufB now = pre-activation of layer 2 (ReLU deferred into out_kernel)

    // ---- output head + softmax
    out_kernel<<<node_blocks, TB, 0, stream>>>(bufB, Wout, bout, out, N);
}
